// Attention_73263552135848
// MI455X (gfx1250) — matrix operand order repaired and checked by reading, not compiled, not run on hardware
//
#include <hip/hip_runtime.h>
#include <hip/hip_bf16.h>

// ---------------------------------------------------------------------------
// MI455X (gfx1250) attention block: QKV proj -> RoPE -> flash attention -> out proj
// Matmuls via v_wmma_f32_16x16x32_bf16; GEMM tiles staged through LDS by the
// Tensor Data Mover (tensor_load_to_lds + s_wait_tensorcnt), double-buffered.
// ---------------------------------------------------------------------------

typedef __bf16 bf16;
typedef __attribute__((ext_vector_type(16))) __bf16 v16bf;
typedef __attribute__((ext_vector_type(8)))  __bf16 v8bf;
typedef __attribute__((ext_vector_type(8)))  float  v8f;
typedef __attribute__((ext_vector_type(4)))  unsigned int u32x4;
typedef __attribute__((ext_vector_type(4)))  int i32x4;
typedef __attribute__((ext_vector_type(8)))  int i32x8;

#define B_    4
#define S_    1024
#define DIM_  2048
#define NH_   32
#define NKV_  8
#define HD_   64
#define EQKV_ 3072   // (NH + 2*NKV) * HD

#if __has_builtin(__builtin_amdgcn_tensor_load_to_lds)
#define HAVE_TDM 1
#else
#define HAVE_TDM 0
#endif

// ---------------- WMMA wrapper ---------------------------------------------
static __device__ __forceinline__ v8f wmma_bf16(v16bf a, v16bf b, v8f c) {
  // (neg_a, A, neg_b, B, c_mod, C, reuse_a, reuse_b)
  return __builtin_amdgcn_wmma_f32_16x16x32_bf16(false, a, false, b, (short)0, c,
                                                 false, false);
}

// A-matrix fragment (16x32, MxK, 16-bit): lane r = lane&15 is the row,
// kh = lane>>4 selects K-half. VGPR 0..3 hold K = kh*8 .. kh*8+7,
// VGPR 4..7 hold K = 16+kh*8 .. 16+kh*8+7  -> two contiguous 16B chunks.
static __device__ __forceinline__ v16bf load_a_frag(const bf16* base, int ld,
                                                    int row0, int k0) {
  int lane = threadIdx.x & 31;
  int r = lane & 15, kh = lane >> 4;
  const bf16* p = base + (size_t)(row0 + r) * ld + k0 + kh * 8;
  v8bf c0 = *(const v8bf*)(p);
  v8bf c1 = *(const v8bf*)(p + 16);
  v16bf f;
#pragma unroll
  for (int i = 0; i < 8; ++i) { f[i] = c0[i]; f[i + 8] = c1[i]; }
  return f;
}

// B-matrix fragment (32x16, KxN, 16-bit): lane n = lane&15 is the column,
// VGPR 0..3 hold K = kh*8 .. kh*8+7, VGPR 4..7 hold K = 16+kh*8 .. 16+kh*8+7
// (same K order as the A fragment) -> two contiguous 16B chunks.
// `col0 + n` indexes rows of the NT-stored matrix W[N,K] (K contiguous).
static __device__ __forceinline__ v16bf load_b_frag(const bf16* base, int ld,
                                                    int col0, int k0) {
  int lane = threadIdx.x & 31;
  int n = lane & 15, kh = lane >> 4;
  const bf16* p = base + (size_t)(col0 + n) * ld + k0 + kh * 8;
  v8bf c0 = *(const v8bf*)(p);
  v8bf c1 = *(const v8bf*)(p + 16);
  v16bf f;
#pragma unroll
  for (int i = 0; i < 8; ++i) { f[i] = c0[i]; f[i + 8] = c1[i]; }
  return f;
}

#if HAVE_TDM
// ---------------- Tensor Data Mover: 2-D bf16 tile -> LDS -------------------
// Builds D# group0/group1 per CDNA5 ISA:
//   group0: count=1, lds_addr, global_addr[56:0], type=2 ("image")
//   group1: data_size=1 (2 bytes), tensor_dim0=ld, tensor_dim1=tile_h,
//           tile_dim0=tile_w, tile_dim1=tile_h, tensor_dim0_stride=ld
// Groups 2/3 are zero (<=2-D tensor). Tracked with TENSORcnt.
static __device__ __forceinline__ void tdm_load_tile(const bf16* gsrc,
                                                     unsigned lds_off,
                                                     int tile_w, int tile_h,
                                                     int ld) {
  unsigned long long ga = (unsigned long long)(uintptr_t)gsrc;
  u32x4 g0 = { 1u,                                   // count=1, user mode
               lds_off,                               // lds_addr (bytes)
               (unsigned)ga,                          // global_addr[31:0]
               (unsigned)((ga >> 32) & 0x1FFFFFFull)  // global_addr[56:32]
                   | (2u << 30) };                    // type=2
  unsigned td0 = (unsigned)ld, td1 = (unsigned)tile_h;
  i32x8 g1 = { (int)(1u << 16),                              // data_size=2B
               (int)((td0 & 0xFFFFu) << 16),                 // tensor_dim0 lo
               (int)((td0 >> 16) | ((td1 & 0xFFFFu) << 16)), // dim0 hi|dim1 lo
               (int)((td1 >> 16) | ((unsigned)tile_w << 16)),// dim1 hi|tile0
               (int)(tile_h & 0xFFFF),                       // tile_dim1
               (int)ld,                                      // dim0_stride lo
               0, 0 };
  i32x4 z4 = { 0, 0, 0, 0 };
#if __clang_major__ >= 23
  i32x8 z8 = { 0, 0, 0, 0, 0, 0, 0, 0 };
  __builtin_amdgcn_tensor_load_to_lds(g0, g1, z4, z4, z8, 0);
#else
  __builtin_amdgcn_tensor_load_to_lds(g0, g1, z4, z4, 0);
#endif
}
#endif  // HAVE_TDM

// ---------------- fp32 -> bf16 conversion ----------------------------------
__global__ void cvt_f32_bf16(const float* __restrict__ in, bf16* __restrict__ out,
                             int n4) {
  int t = blockIdx.x * blockDim.x + threadIdx.x;
  if (t >= n4) return;
  const float4 v = ((const float4*)in)[t];
  int o = t * 4;
  out[o + 0] = (bf16)v.x;
  out[o + 1] = (bf16)v.y;
  out[o + 2] = (bf16)v.z;
  out[o + 3] = (bf16)v.w;
}

// ---------------- NT GEMM: C[M,N] = A[M,K] * W[N,K]^T ----------------------
// 256 threads = 8 waves; block tile 128(M) x 64(N); each wave 32x32 (2x2 WMMA).
// TDM double-buffers 128x32 A-slabs and 64x32 W-slabs in LDS so the DMA of
// slab k+32 overlaps the WMMA stage on slab k (TENSORcnt pipelining).
template <typename OutT>
__global__ __launch_bounds__(256) void gemm_nt(const bf16* __restrict__ A,
                                               const bf16* __restrict__ W,
                                               OutT* __restrict__ C,
                                               int M, int N, int K) {
  int wave = threadIdx.x >> 5;
  int lane = threadIdx.x & 31;
  int mblk = blockIdx.x * 128;
  int nblk = blockIdx.y * 64;
  int mw = (wave >> 1) * 32;  // wave row offset inside block tile
  int nw = (wave & 1) * 32;   // wave col offset inside block tile

  v8f acc[2][2] = {};

#if HAVE_TDM
  __shared__ __align__(128) bf16 ldsA[2][128 * 32];
  __shared__ __align__(128) bf16 ldsB[2][64 * 32];
  const bf16* gA = A + (size_t)mblk * K;
  const bf16* gW = W + (size_t)nblk * K;

  int cur = 0;
  if (threadIdx.x < 32) {  // wave 0 drives the TDM for the whole block
    tdm_load_tile(gA, (unsigned)(uintptr_t)&ldsA[0][0], 32, 128, K);
    tdm_load_tile(gW, (unsigned)(uintptr_t)&ldsB[0][0], 32, 64, K);
  }
  for (int k = 0; k < K; k += 32) {
    if (threadIdx.x < 32) {
      if (k + 32 < K) {  // prefetch next slab into the other buffer
        tdm_load_tile(gA + (k + 32), (unsigned)(uintptr_t)&ldsA[cur ^ 1][0],
                      32, 128, K);
        tdm_load_tile(gW + (k + 32), (unsigned)(uintptr_t)&ldsB[cur ^ 1][0],
                      32, 64, K);
        __builtin_amdgcn_s_wait_tensorcnt(2);  // current buffer's 2 ops done
      } else {
        __builtin_amdgcn_s_wait_tensorcnt(0);
      }
    }
    __syncthreads();  // slab `cur` visible to all waves
    v16bf a0 = load_a_frag(ldsA[cur], 32, mw, 0);
    v16bf a1 = load_a_frag(ldsA[cur], 32, mw + 16, 0);
    v16bf b0 = load_b_frag(ldsB[cur], 32, nw, 0);
    v16bf b1 = load_b_frag(ldsB[cur], 32, nw + 16, 0);
    acc[0][0] = wmma_bf16(a0, b0, acc[0][0]);
    acc[0][1] = wmma_bf16(a0, b1, acc[0][1]);
    acc[1][0] = wmma_bf16(a1, b0, acc[1][0]);
    acc[1][1] = wmma_bf16(a1, b1, acc[1][1]);
    __syncthreads();  // all waves done with slab `cur` before it is re-filled
    cur ^= 1;
  }
#else
  // Fallback: direct global fragment loads (L2-resident working set).
  for (int k = 0; k < K; k += 32) {
    if (k + 32 < K) {
      __builtin_prefetch(A + (size_t)(mblk + mw + lane) * K + k + 32, 0, 0);
      __builtin_prefetch(W + (size_t)(nblk + nw + lane) * K + k + 32, 0, 0);
    }
    v16bf a0 = load_a_frag(A, K, mblk + mw, k);
    v16bf a1 = load_a_frag(A, K, mblk + mw + 16, k);
    v16bf b0 = load_b_frag(W, K, nblk + nw, k);
    v16bf b1 = load_b_frag(W, K, nblk + nw + 16, k);
    acc[0][0] = wmma_bf16(a0, b0, acc[0][0]);
    acc[0][1] = wmma_bf16(a0, b1, acc[0][1]);
    acc[1][0] = wmma_bf16(a1, b0, acc[1][0]);
    acc[1][1] = wmma_bf16(a1, b1, acc[1][1]);
  }
#endif

  // C/D layout: lane n = lane&15 is column; VGPR i is row i + 8*(lane>>4).
  int n = lane & 15, hi = lane >> 4;
#pragma unroll
  for (int ti = 0; ti < 2; ++ti)
#pragma unroll
    for (int tj = 0; tj < 2; ++tj)
#pragma unroll
      for (int i = 0; i < 8; ++i) {
        int row = mblk + mw + ti * 16 + i + 8 * hi;
        int col = nblk + nw + tj * 16 + n;
        C[(size_t)row * N + col] = (OutT)acc[ti][tj][i];
      }
}

// ---------------- RoPE + head split (q scaled by 1/sqrt(HD)) ---------------
// qkv[b,s,e] bf16 -> qb [b,h,s,d], kb [b,kv,s,d], vt [b,kv,d,s] (V transposed)
__global__ void rope_split(const bf16* __restrict__ qkv, const float* __restrict__ fc,
                           bf16* __restrict__ qb, bf16* __restrict__ kb,
                           bf16* __restrict__ vt) {
  int t = blockIdx.x * blockDim.x + threadIdx.x;
  int pair = t & 31;
  int rest = t >> 5;
  int head = rest % (NH_ + 2 * NKV_);
  int tok  = rest / (NH_ + 2 * NKV_);
  if (tok >= B_ * S_) return;
  int b = tok / S_, s = tok % S_;

  size_t src = (size_t)tok * EQKV_ + head * HD_ + 2 * pair;
  float x0 = (float)qkv[src];
  float x1 = (float)qkv[src + 1];

  if (head < NH_ + NKV_) {
    float c  = fc[s * HD_ + 2 * pair];
    float sn = fc[s * HD_ + 2 * pair + 1];
    float r0 = x0 * c - x1 * sn;
    float r1 = x1 * c + x0 * sn;
    if (head < NH_) {
      const float sc = 0.125f;  // 1/sqrt(64) folded into q
      size_t o = ((size_t)((b * NH_ + head) * S_ + s)) * HD_ + 2 * pair;
      qb[o] = (bf16)(r0 * sc);
      qb[o + 1] = (bf16)(r1 * sc);
    } else {
      int kv = head - NH_;
      size_t o = ((size_t)((b * NKV_ + kv) * S_ + s)) * HD_ + 2 * pair;
      kb[o] = (bf16)r0;
      kb[o + 1] = (bf16)r1;
    }
  } else {
    int kv = head - (NH_ + NKV_);
    size_t o = ((size_t)((b * NKV_ + kv) * HD_ + 2 * pair)) * S_ + s;
    vt[o] = (bf16)x0;        // dim 2*pair
    vt[o + S_] = (bf16)x1;   // dim 2*pair+1
  }
}

// ---------------- Flash attention (causal, GQA 4:1) ------------------------
// Block = 128 threads = 4 waves; each wave owns a 16-query tile of one (b,h).
__global__ __launch_bounds__(128) void attn_fwd(const bf16* __restrict__ qb,
                                                const bf16* __restrict__ kb,
                                                const bf16* __restrict__ vt,
                                                bf16* __restrict__ yb) {
  __shared__ __align__(64) bf16 lsp[4 * 16 * 32];  // per-wave 16x32 P tile

  int wave = threadIdx.x >> 5, lane = threadIdx.x & 31;
  int n = lane & 15, hi = lane >> 4;
  int b = blockIdx.z, h = blockIdx.y;
  int q0 = blockIdx.x * 64 + wave * 16;
  int kv = h >> 2;  // NH/NKV = 4

  const bf16* Q  = qb + (size_t)(b * NH_ + h) * S_ * HD_;
  const bf16* Kp = kb + (size_t)(b * NKV_ + kv) * S_ * HD_;
  const bf16* Vp = vt + (size_t)(b * NKV_ + kv) * HD_ * S_;
  bf16* lp = lsp + wave * (16 * 32);

  // Q fragments live in registers for the whole K loop (HD=64 -> two A frags)
  v16bf aq0 = load_a_frag(Q, HD_, q0, 0);
  v16bf aq1 = load_a_frag(Q, HD_, q0, 32);

  v8f acc[4] = {};
  float mrow[8], lrow[8];
#pragma unroll
  for (int i = 0; i < 8; ++i) { mrow[i] = -1e30f; lrow[i] = 0.f; }

  for (int j = 0; j < q0 + 16; j += 32) {  // 32-key chunks, causal bound
    // scores: 16 queries x 32 keys = two 16x16 WMMA tiles, K-dim = 64
    v8f s0 = {}, s1 = {};
    s0 = wmma_bf16(aq0, load_b_frag(Kp, HD_, j, 0), s0);
    s0 = wmma_bf16(aq1, load_b_frag(Kp, HD_, j, 32), s0);
    s1 = wmma_bf16(aq0, load_b_frag(Kp, HD_, j + 16, 0), s1);
    s1 = wmma_bf16(aq1, load_b_frag(Kp, HD_, j + 16, 32), s1);

    if (j + 31 > q0) {  // diagonal chunk: apply causal mask elementwise
#pragma unroll
      for (int i = 0; i < 8; ++i) {
        int q = q0 + i + 8 * hi;
        if (j + n > q)      s0[i] = -1e30f;
        if (j + 16 + n > q) s1[i] = -1e30f;
      }
    }

    // online softmax: rows live as (VGPR i, 16-lane half) -> width-16 butterflies
    float alpha[8];
#pragma unroll
    for (int i = 0; i < 8; ++i) {
      float t = fmaxf(s0[i], s1[i]);
#pragma unroll
      for (int off = 1; off < 16; off <<= 1) t = fmaxf(t, __shfl_xor(t, off, 16));
      float mn = fmaxf(mrow[i], t);
      alpha[i] = __expf(mrow[i] - mn);
      mrow[i] = mn;
      float p0 = __expf(s0[i] - mn);
      float p1 = __expf(s1[i] - mn);
      s0[i] = p0; s1[i] = p1;
      float rs = p0 + p1;
#pragma unroll
      for (int off = 1; off < 16; off <<= 1) rs += __shfl_xor(rs, off, 16);
      lrow[i] = lrow[i] * alpha[i] + rs;
    }

    // stage P (C layout) into per-wave LDS as a row-major 16x32 bf16 A-matrix
#pragma unroll
    for (int i = 0; i < 8; ++i) {
      int row = i + 8 * hi;
      lp[row * 32 + n]      = (bf16)s0[i];
      lp[row * 32 + 16 + n] = (bf16)s1[i];
    }
    asm volatile("s_wait_dscnt 0" ::: "memory");
    v16bf ap = load_a_frag(lp, 32, 0, 0);

    // P (16x32) x V (32 keys x 64 dims) — V transposed, contiguous B frags
#pragma unroll
    for (int t = 0; t < 4; ++t) {
      v16bf bv = load_b_frag(Vp, S_, t * 16, j);
      v8f c = acc[t];
#pragma unroll
      for (int i = 0; i < 8; ++i) c[i] *= alpha[i];
      acc[t] = wmma_bf16(ap, bv, c);
    }
  }

  // normalize and store y[b, s, h*64+d] (bf16) for the output projection
  float rl[8];
#pragma unroll
  for (int i = 0; i < 8; ++i) rl[i] = 1.0f / lrow[i];
#pragma unroll
  for (int t = 0; t < 4; ++t)
#pragma unroll
    for (int i = 0; i < 8; ++i) {
      int q = q0 + i + 8 * hi;
      size_t o = ((size_t)(b * S_ + q)) * DIM_ + h * HD_ + t * 16 + n;
      yb[o] = (bf16)(acc[t][i] * rl[i]);
    }
}

// ---------------------------------------------------------------------------
extern "C" void kernel_launch(void* const* d_in, const int* in_sizes, int n_in,
                              void* d_out, int out_size, void* d_ws, size_t ws_size,
                              hipStream_t stream) {
  const float* x     = (const float*)d_in[0];
  const float* freqs = (const float*)d_in[1];
  // d_in[2] = causal mask: recomputed analytically in-kernel, unused
  const float* wqkv  = (const float*)d_in[3];
  const float* wo    = (const float*)d_in[4];
  float* out = (float*)d_out;

  char* ws = (char*)d_ws;
  // 256B-aligned workspace carve-out (bf16 buffers), total ~100 MB
  bf16* xb    = (bf16*)(ws + 0);          //  4096x2048          16.0 MB
  bf16* wqb   = (bf16*)(ws + 16777216);   //  3072x2048          12.0 MB
  bf16* wob   = (bf16*)(ws + 29360128);   //  2048x2048           8.0 MB
  bf16* qkvb  = (bf16*)(ws + 37748736);   //  4096x3072          24.0 MB
  bf16* qrb   = (bf16*)(ws + 62914560);   //  [4,32,1024,64]     16.0 MB
  bf16* krb   = (bf16*)(ws + 79691776);   //  [4,8,1024,64]       4.0 MB
  bf16* vtb   = (bf16*)(ws + 83886080);   //  [4,8,64,1024]       4.0 MB
  bf16* yb    = (bf16*)(ws + 88080384);   //  4096x2048          16.0 MB

  const int M = B_ * S_;  // 4096

  // 1) downcast inputs to bf16
  cvt_f32_bf16<<<(M * DIM_ / 4 + 255) / 256, 256, 0, stream>>>(x, xb, M * DIM_ / 4);
  cvt_f32_bf16<<<(EQKV_ * DIM_ / 4 + 255) / 256, 256, 0, stream>>>(wqkv, wqb,
                                                                   EQKV_ * DIM_ / 4);
  cvt_f32_bf16<<<(DIM_ * DIM_ / 4 + 255) / 256, 256, 0, stream>>>(wo, wob,
                                                                  DIM_ * DIM_ / 4);

  // 2) qkv = x . wqkv^T   (4096 x 2048 x 3072)
  gemm_nt<bf16><<<dim3(M / 128, EQKV_ / 64), 256, 0, stream>>>(xb, wqb, qkvb,
                                                               M, EQKV_, DIM_);

  // 3) RoPE + split into q (scaled), k, v(transposed)
  {
    int total = M * (NH_ + 2 * NKV_) * (HD_ / 2);
    rope_split<<<(total + 255) / 256, 256, 0, stream>>>(qkvb, freqs, qrb, krb, vtb);
  }

  // 4) causal flash attention, GQA 4:1
  attn_fwd<<<dim3(S_ / 64, NH_, B_), 128, 0, stream>>>(qrb, krb, vtb, yb);

  // 5) out = y . wo^T   (4096 x 2048 x 2048), fp32 result
  gemm_nt<float><<<dim3(M / 128, DIM_ / 64), 256, 0, stream>>>(yb, wob, out,
                                                               M, DIM_, DIM_);
}
